// NestedBlock_38345468018691
// MI455X (gfx1250) — matrix operand order repaired and checked by reading, not compiled, not run on hardware
//
#include <hip/hip_runtime.h>
#include <hip/hip_bf16.h>
#include <cmath>

typedef __bf16 bf16;
typedef __attribute__((ext_vector_type(8)))  __bf16 v8bf;
typedef __attribute__((ext_vector_type(16))) __bf16 v16bf;
typedef __attribute__((ext_vector_type(8)))  float  v8f;
typedef __attribute__((ext_vector_type(4)))  int    nb_v4i;

#define DIMF 768
#define NTOK 1024
#define BATCH 4
#define NHEAD 12
#define HD 64

#if defined(__gfx1250__) && __has_builtin(__builtin_amdgcn_global_load_async_to_lds_b128)
#define HAVE_ASYNC_LDS 1
#else
#define HAVE_ASYNC_LDS 0
#endif

__device__ __forceinline__ bf16 tobf(float f) {
  unsigned u = __float_as_uint(f);
  unsigned r = u + 0x7FFFu + ((u >> 16) & 1u);
  unsigned short s = (unsigned short)(r >> 16);
  return __builtin_bit_cast(bf16, s);
}

__device__ __forceinline__ v16bf mkfrag(const bf16* lo, const bf16* hi) {
  v8bf a = *(const v8bf*)lo;
  v8bf b = *(const v8bf*)hi;
  return __builtin_shufflevector(a, b, 0,1,2,3,4,5,6,7,8,9,10,11,12,13,14,15);
}

__device__ __forceinline__ v8f wmma_bf(v16bf a, v16bf b, v8f c) {
  return __builtin_amdgcn_wmma_f32_16x16x32_bf16(false, a, false, b, (short)0, c, false, false);
}

// Async global->LDS copy of 16 bytes (8 bf16) per lane; ASYNCcnt-tracked.
__device__ __forceinline__ void async_copy16(const bf16* g, bf16* l) {
#if HAVE_ASYNC_LDS
  nb_v4i* gg = (nb_v4i*)g;   // const_cast + reinterpret in generic AS
  nb_v4i* ll = (nb_v4i*)l;
  __builtin_amdgcn_global_load_async_to_lds_b128(
      (__attribute__((address_space(1))) nb_v4i*)gg,
      (__attribute__((address_space(3))) nb_v4i*)ll, 0, 0);
#else
  *(v8bf*)l = *(const v8bf*)g;
#endif
}

__device__ __forceinline__ void async_wait() {
#if HAVE_ASYNC_LDS
#if __has_builtin(__builtin_amdgcn_s_wait_asynccnt)
  __builtin_amdgcn_s_wait_asynccnt(0);
#else
  asm volatile("s_wait_asynccnt 0" ::: "memory");
#endif
#endif
}

// ---------------------------------------------------------------------------
// Kernel 1: expert-preferred capacity router (1 block per batch, 1024 thr)
// ---------------------------------------------------------------------------
__global__ __launch_bounds__(1024) void nb_router(
    const float* __restrict__ x, const float* __restrict__ rw,
    const float* __restrict__ rb, float* __restrict__ rprob,
    int* __restrict__ dtok) {
  __shared__ float key[1024];
  __shared__ int   idx[1024];
  __shared__ int   flag[1024];
  int b = blockIdx.x, n = threadIdx.x;
  const float* xr = x + ((size_t)b * NTOK + n) * DIMF;
  float a0 = rb[0], a1 = rb[1], a2 = rb[2], a3 = rb[3];
  for (int d = 0; d < DIMF; ++d) {
    float xv = xr[d];
    a0 += xv * rw[d*4+0]; a1 += xv * rw[d*4+1];
    a2 += xv * rw[d*4+2]; a3 += xv * rw[d*4+3];
  }
  float mx = fmaxf(fmaxf(a0, a1), fmaxf(a2, a3));
  float p[4];
  p[0]=__expf(a0-mx); p[1]=__expf(a1-mx); p[2]=__expf(a2-mx); p[3]=__expf(a3-mx);
  float inv = 1.0f / (p[0]+p[1]+p[2]+p[3]);
  p[0]*=inv; p[1]*=inv; p[2]*=inv; p[3]*=inv;
  const int ks[4] = {409, 307, 204, 104};  // 0.4/0.3/0.2/0.1*1024, remainder->e3
  bool avail = true; int assigned = 0;
  for (int e = 3; e >= 0; --e) {
    key[n] = avail ? p[e] : -3.0e38f;
    idx[n] = n;
    flag[n] = 0;
    __syncthreads();
    for (int k = 2; k <= 1024; k <<= 1)
      for (int j = k >> 1; j > 0; j >>= 1) {
        int ixj = n ^ j;
        if (ixj > n) {
          float k0 = key[n], k1 = key[ixj];
          int   i0 = idx[n], i1 = idx[ixj];
          bool firstGE = (k0 > k1) || (k0 == k1 && i0 < i1);
          bool desc = ((n & k) == 0);
          if (desc != firstGE) { key[n]=k1; key[ixj]=k0; idx[n]=i1; idx[ixj]=i0; }
        }
        __syncthreads();
      }
    if (n < ks[e]) flag[idx[n]] = 1;
    __syncthreads();
    if (flag[n] && avail) { assigned = e; avail = false; }
    __syncthreads();
  }
  rprob[b*NTOK + n] = p[assigned];
  dtok[b*NTOK + n]  = DIMF >> (3 - assigned);
}

// ---------------------------------------------------------------------------
// Kernel 2/6: LayerNorm (-> bf16), optionally masked to nested dims
// ---------------------------------------------------------------------------
template<bool MASKED>
__global__ __launch_bounds__(256) void nb_layernorm(
    const float* __restrict__ x, const float* __restrict__ g,
    const float* __restrict__ bta, const int* __restrict__ dtok,
    bf16* __restrict__ out) {
  __shared__ float sred[8];
  int row = blockIdx.x, t = threadIdx.x;
  int lane = t & 31, w = t >> 5;
  const float* xr = x + (size_t)row * DIMF;
  float v0 = xr[t], v1 = xr[t+256], v2 = xr[t+512];
  float s = v0 + v1 + v2;
  #pragma unroll
  for (int m = 16; m >= 1; m >>= 1) s += __shfl_xor(s, m, 32);
  if (lane == 0) sred[w] = s;
  __syncthreads();
  if (t < 8) {
    float v = sred[t];
    #pragma unroll
    for (int m = 4; m >= 1; m >>= 1) v += __shfl_xor(v, m, 32);
    if (t == 0) sred[0] = v;
  }
  __syncthreads();
  float mu = sred[0] * (1.0f / DIMF);
  __syncthreads();
  float d0 = v0-mu, d1 = v1-mu, d2 = v2-mu;
  float q = d0*d0 + d1*d1 + d2*d2;
  #pragma unroll
  for (int m = 16; m >= 1; m >>= 1) q += __shfl_xor(q, m, 32);
  if (lane == 0) sred[w] = q;
  __syncthreads();
  if (t < 8) {
    float v = sred[t];
    #pragma unroll
    for (int m = 4; m >= 1; m >>= 1) v += __shfl_xor(v, m, 32);
    if (t == 0) sred[0] = v;
  }
  __syncthreads();
  float rstd = rsqrtf(sred[0] * (1.0f / DIMF) + 1e-5f);
  int dt = MASKED ? dtok[row] : DIMF;
  bf16* orow = out + (size_t)row * DIMF;
  float y0 = d0*rstd*g[t]     + bta[t];
  float y1 = d1*rstd*g[t+256] + bta[t+256];
  float y2 = d2*rstd*g[t+512] + bta[t+512];
  orow[t]     = tobf((t       < dt) ? y0 : 0.0f);
  orow[t+256] = tobf((t + 256 < dt) ? y1 : 0.0f);
  orow[t+512] = tobf((t + 512 < dt) ? y2 : 0.0f);
}

// ---------------------------------------------------------------------------
// WMMA GEMM: C[M,N] = A(bf16)[M,K] @ W(f32->bf16)[K,N]; 128x128x32 tiles,
// 8 waves, wave tile 64x32 (4x2 of 16x16), v_wmma_f32_16x16x32_bf16.
// A tile staged via GLOBAL_LOAD_ASYNC_TO_LDS (ASYNCcnt); B tile converted
// f32->bf16 and transposed at LDS-store time.
// EP: 0=QKV (bias+nested-mask -> bf16), 1=WO (bias+mask+residual -> f32),
//     2=W1 (bias+GELU -> bf16), 3=W2 (bias, z + rprob*val -> f32)
// ---------------------------------------------------------------------------
template<int EP>
__global__ __launch_bounds__(256) void nb_gemm(
    const bf16* __restrict__ A, const float* __restrict__ W,
    const float* __restrict__ bias, bf16* __restrict__ outb,
    float* __restrict__ outf, const float* __restrict__ resid,
    const float* __restrict__ rprob, const int* __restrict__ dtok,
    int M, int Nfull, int K) {
  __shared__ bf16 As[128 * 40];   // [m][k], stride 40 (16B-aligned frag reads)
  __shared__ bf16 Bt[128 * 40];   // [n][k], transposed at store time
  int t = threadIdx.x;
  int bn = blockIdx.x * 128, bm = blockIdx.y * 128;
  int lane = t & 31, wave = t >> 5;
  int wm = wave >> 2, wn = wave & 3;
  int hh = lane >> 4, ln = lane & 15;
  const v8f vzero = {0.f,0.f,0.f,0.f,0.f,0.f,0.f,0.f};
  v8f acc[4][2];
  #pragma unroll
  for (int i = 0; i < 4; ++i) { acc[i][0] = vzero; acc[i][1] = vzero; }

  int arow = t >> 1, akh = (t & 1) * 16;        // A: 2 thr/row, 16 bf16 each
  int bkr  = t >> 3, bn0 = (t & 7) * 16;        // B: 8 thr/k-row, 16 f32 each
  const bf16*  aptr = A + (size_t)(bm + arow) * K + akh;
  const float* wptr = W + (size_t)bkr * Nfull + bn + bn0;
  bf16* asl = &As[arow * 40 + akh];

  for (int k0 = 0; k0 < K; k0 += 32) {
    const float* wk = wptr + (size_t)k0 * Nfull;
    float4 f0 = *(const float4*)(wk);
    float4 f1 = *(const float4*)(wk + 4);
    float4 f2 = *(const float4*)(wk + 8);
    float4 f3 = *(const float4*)(wk + 12);
    if (k0 + 32 < K) {
      __builtin_prefetch(aptr + k0 + 32, 0, 1);
      __builtin_prefetch(wk + (size_t)32 * Nfull, 0, 1);
    }
    __syncthreads();
    // A tile: async DMA straight into LDS (no VGPR round-trip)
    async_copy16(aptr + k0, asl);
    async_copy16(aptr + k0 + 8, asl + 8);
    // B tile: f32 -> bf16 convert + transpose
    float wv[16] __attribute__((aligned(16)));
    *(float4*)(wv + 0) = f0; *(float4*)(wv + 4)  = f1;
    *(float4*)(wv + 8) = f2; *(float4*)(wv + 12) = f3;
    #pragma unroll
    for (int i = 0; i < 16; ++i) Bt[(bn0 + i) * 40 + bkr] = tobf(wv[i]);
    async_wait();
    __syncthreads();
    // B frags: lane -> col n = ln, K halves via hh
    const bf16* br0 = &Bt[(wn * 32 + ln) * 40 + hh * 16];
    const bf16* br1 = &Bt[(wn * 32 + 16 + ln) * 40 + hh * 16];
    v16bf bf0 = mkfrag(br0, br0 + 8);
    v16bf bf1 = mkfrag(br1, br1 + 8);
    #pragma unroll
    for (int mt = 0; mt < 4; ++mt) {
      const bf16* ar = &As[(wm * 64 + mt * 16 + ln) * 40];
      v16bf af = mkfrag(ar + 16 * hh, ar + 16 * hh + 8);
      acc[mt][0] = wmma_bf(af, bf0, acc[mt][0]);
      acc[mt][1] = wmma_bf(af, bf1, acc[mt][1]);
    }
  }
  #pragma unroll
  for (int mt = 0; mt < 4; ++mt)
    #pragma unroll
    for (int nt = 0; nt < 2; ++nt) {
      int col = bn + wn * 32 + nt * 16 + ln;
      float bv = bias[col];
      v8f v = acc[mt][nt];
      #pragma unroll
      for (int r = 0; r < 8; ++r) {
        int row = bm + wm * 64 + mt * 16 + r + 8 * hh;
        float val = v[r] + bv;
        if constexpr (EP == 0) {
          int f = col % DIMF;
          outb[(size_t)row * Nfull + col] = tobf((f < dtok[row]) ? val : 0.0f);
        } else if constexpr (EP == 1) {
          float mv = (col < dtok[row]) ? val : 0.0f;
          outf[(size_t)row * Nfull + col] =
              resid[(size_t)row * Nfull + col] + mv;
        } else if constexpr (EP == 2) {
          float gl = 0.5f * val * (1.0f + erff(val * 0.70710678118f));
          outb[(size_t)row * Nfull + col] = tobf(gl);
        } else {
          outf[(size_t)row * Nfull + col] =
              resid[(size_t)row * Nfull + col] + rprob[row] * val;
        }
      }
    }
}

// ---------------------------------------------------------------------------
// Flash attention: one block per (qblk of 64 rows, head, batch). 4 waves,
// each wave owns 16 q-rows; 32-key chunks; softmax stats in C-layout.
// Q/K tiles staged via async LDS DMA; V transposed manually.
// ---------------------------------------------------------------------------
__global__ __launch_bounds__(128) void nb_attention(
    const bf16* __restrict__ qkv, bf16* __restrict__ attn,
    const int* __restrict__ dtok) {
  __shared__ bf16 Qs[64 * 64];
  __shared__ bf16 Ks[32 * 64];
  __shared__ bf16 Vt[64 * 40];        // [hd][key], stride 40
  __shared__ bf16 Ps[4][16 * 32];     // per-wave P staging
  int t = threadIdx.x;
  int qblk = blockIdx.x, h = blockIdx.y, b = blockIdx.z;
  int rowb = b * NTOK, qbase = qblk * 64;
  int lane = t & 31, wave = t >> 5, hh = lane >> 4, ln = lane & 15;
  int r4 = t >> 2, c0 = (t & 3) * 16;
  #pragma unroll
  for (int p = 0; p < 2; ++p) {
    int r = p * 32 + r4;
    const bf16* src = qkv + (size_t)(rowb + qbase + r) * 2304 + h * HD + c0;
    async_copy16(src, &Qs[r * 64 + c0]);
    async_copy16(src + 8, &Qs[r * 64 + c0 + 8]);
  }
  async_wait();
  __syncthreads();
  int mrow = wave * 16 + ln;
  v16bf qf0 = mkfrag(&Qs[mrow*64 + 16*hh],      &Qs[mrow*64 + 8 + 16*hh]);
  v16bf qf1 = mkfrag(&Qs[mrow*64 + 32 + 16*hh], &Qs[mrow*64 + 40 + 16*hh]);

  const v8f vzero = {0.f,0.f,0.f,0.f,0.f,0.f,0.f,0.f};
  v8f O0 = vzero, O1 = vzero, O2 = vzero, O3 = vzero;
  float mrw[8], lrw[8];
  #pragma unroll
  for (int r = 0; r < 8; ++r) { mrw[r] = -3.0e38f; lrw[r] = 0.0f; }

  for (int j = 0; j < NTOK / 32; ++j) {
    int kb = j * 32;
    const bf16* ksrc = qkv + (size_t)(rowb + kb + r4) * 2304 + DIMF   + h*HD + c0;
    const bf16* vsrc = qkv + (size_t)(rowb + kb + r4) * 2304 + 2*DIMF + h*HD + c0;
    v8bf vl0 = *(const v8bf*)vsrc, vl1 = *(const v8bf*)(vsrc + 8);
    if (j + 1 < NTOK / 32) __builtin_prefetch(ksrc + (size_t)32 * 2304, 0, 1);
    __syncthreads();
    async_copy16(ksrc, &Ks[r4 * 64 + c0]);
    async_copy16(ksrc + 8, &Ks[r4 * 64 + c0 + 8]);
    bf16 vtmp[16] __attribute__((aligned(16)));
    *(v8bf*)vtmp = vl0; *(v8bf*)(vtmp + 8) = vl1;
    #pragma unroll
    for (int i = 0; i < 16; ++i) Vt[(c0 + i) * 40 + r4] = vtmp[i];  // transpose
    async_wait();
    __syncthreads();
    // scores for two 16-key tiles (K = hd = 64 -> 2 WMMA k-steps each)
    v8f s0 = vzero, s1 = vzero;
    const bf16* kr0  = &Ks[ln * 64 + 16 * hh];
    const bf16* kr0b = &Ks[ln * 64 + 32 + 16 * hh];
    s0 = wmma_bf(qf0, mkfrag(kr0, kr0 + 8), s0);
    s0 = wmma_bf(qf1, mkfrag(kr0b, kr0b + 8), s0);
    const bf16* kr1  = &Ks[(16 + ln) * 64 + 16 * hh];
    const bf16* kr1b = &Ks[(16 + ln) * 64 + 32 + 16 * hh];
    s1 = wmma_bf(qf0, mkfrag(kr1, kr1 + 8), s1);
    s1 = wmma_bf(qf1, mkfrag(kr1b, kr1b + 8), s1);
    // online softmax; row r+8*hh reduced across the 16-lane half
    float aarr[8];
    #pragma unroll
    for (int r = 0; r < 8; ++r) {
      float sa = s0[r] * 0.125f, sb = s1[r] * 0.125f;
      float tm = fmaxf(sa, sb);
      tm = fmaxf(tm, __shfl_xor(tm, 1, 32));
      tm = fmaxf(tm, __shfl_xor(tm, 2, 32));
      tm = fmaxf(tm, __shfl_xor(tm, 4, 32));
      tm = fmaxf(tm, __shfl_xor(tm, 8, 32));
      float mn = fmaxf(mrw[r], tm);
      float alpha = __expf(mrw[r] - mn);
      mrw[r] = mn;
      float pa = __expf(sa - mn), pb = __expf(sb - mn);
      float ps = pa + pb;
      ps += __shfl_xor(ps, 1, 32);
      ps += __shfl_xor(ps, 2, 32);
      ps += __shfl_xor(ps, 4, 32);
      ps += __shfl_xor(ps, 8, 32);
      lrw[r] = lrw[r] * alpha + ps;
      aarr[r] = alpha;
      int prow = r + 8 * hh;
      Ps[wave][prow * 32 + ln]      = tobf(pa);
      Ps[wave][prow * 32 + 16 + ln] = tobf(pb);
    }
    asm volatile("s_wait_dscnt 0" ::: "memory");  // wave-local LDS RAW
    const bf16* pr = &Ps[wave][ln * 32];
    v16bf pf = mkfrag(pr + 16 * hh, pr + 16 * hh + 8);
    #pragma unroll
    for (int r = 0; r < 8; ++r) {
      O0[r] *= aarr[r]; O1[r] *= aarr[r]; O2[r] *= aarr[r]; O3[r] *= aarr[r];
    }
    const bf16* v0p = &Vt[(0*16 + ln) * 40 + 16 * hh];
    const bf16* v1p = &Vt[(1*16 + ln) * 40 + 16 * hh];
    const bf16* v2p = &Vt[(2*16 + ln) * 40 + 16 * hh];
    const bf16* v3p = &Vt[(3*16 + ln) * 40 + 16 * hh];
    O0 = wmma_bf(pf, mkfrag(v0p, v0p + 8), O0);
    O1 = wmma_bf(pf, mkfrag(v1p, v1p + 8), O1);
    O2 = wmma_bf(pf, mkfrag(v2p, v2p + 8), O2);
    O3 = wmma_bf(pf, mkfrag(v3p, v3p + 8), O3);
  }
  #pragma unroll
  for (int r = 0; r < 8; ++r) {
    int grow = rowb + qbase + wave * 16 + r + 8 * hh;
    int dt = dtok[grow];
    float invl = 1.0f / lrw[r];
    bf16* orow = attn + (size_t)grow * DIMF + h * HD;
    int fb = h * HD + ln;
    orow[ln]      = tobf((fb      < dt) ? O0[r] * invl : 0.0f);
    orow[16 + ln] = tobf((fb + 16 < dt) ? O1[r] * invl : 0.0f);
    orow[32 + ln] = tobf((fb + 32 < dt) ? O2[r] * invl : 0.0f);
    orow[48 + ln] = tobf((fb + 48 < dt) ? O3[r] * invl : 0.0f);
  }
}

// ---------------------------------------------------------------------------
extern "C" void kernel_launch(void* const* d_in, const int* in_sizes, int n_in,
                              void* d_out, int out_size, void* d_ws, size_t ws_size,
                              hipStream_t stream) {
  (void)in_sizes; (void)n_in; (void)out_size; (void)ws_size;
  const float* x    = (const float*)d_in[0];
  const float* g1   = (const float*)d_in[1];
  const float* bg1  = (const float*)d_in[2];
  const float* rw   = (const float*)d_in[3];
  const float* rb   = (const float*)d_in[4];
  const float* wqkv = (const float*)d_in[5];
  const float* bqkv = (const float*)d_in[6];
  const float* wo   = (const float*)d_in[7];
  const float* bo   = (const float*)d_in[8];
  const float* g2   = (const float*)d_in[9];
  const float* bg2  = (const float*)d_in[10];
  const float* w1   = (const float*)d_in[11];
  const float* b1   = (const float*)d_in[12];
  const float* w2   = (const float*)d_in[13];
  const float* b2   = (const float*)d_in[14];
  float* out = (float*)d_out;

  char* ws = (char*)d_ws;
  size_t off = 0;
  auto alloc = [&](size_t bytes) {
    char* p = ws + off;
    off += (bytes + 255) & ~(size_t)255;
    return p;
  };
  const int M = BATCH * NTOK;  // 4096
  int*   dtok  = (int*)  alloc((size_t)M * 4);
  float* rprob = (float*)alloc((size_t)M * 4);
  bf16*  hm    = (bf16*) alloc((size_t)M * DIMF * 2);
  bf16*  qkvb  = (bf16*) alloc((size_t)M * 3 * DIMF * 2);
  bf16*  attnb = (bf16*) alloc((size_t)M * DIMF * 2);
  float* z     = (float*)alloc((size_t)M * DIMF * 4);
  bf16*  h2    = (bf16*) alloc((size_t)M * DIMF * 2);
  bf16*  a1    = (bf16*) alloc((size_t)M * 4 * DIMF * 2);

  nb_router<<<dim3(BATCH), dim3(1024), 0, stream>>>(x, rw, rb, rprob, dtok);
  nb_layernorm<true><<<dim3(M), dim3(256), 0, stream>>>(x, g1, bg1, dtok, hm);
  nb_gemm<0><<<dim3(2304/128, M/128), dim3(256), 0, stream>>>(
      hm, wqkv, bqkv, qkvb, nullptr, nullptr, nullptr, dtok, M, 2304, 768);
  nb_attention<<<dim3(NTOK/64, NHEAD, BATCH), dim3(128), 0, stream>>>(
      qkvb, attnb, dtok);
  nb_gemm<1><<<dim3(768/128, M/128), dim3(256), 0, stream>>>(
      attnb, wo, bo, nullptr, z, x, nullptr, dtok, M, 768, 768);
  nb_layernorm<false><<<dim3(M), dim3(256), 0, stream>>>(z, g2, bg2, nullptr, h2);
  nb_gemm<2><<<dim3(3072/128, M/128), dim3(256), 0, stream>>>(
      h2, w1, b1, a1, nullptr, nullptr, nullptr, nullptr, M, 3072, 768);
  nb_gemm<3><<<dim3(768/128, M/128), dim3(256), 0, stream>>>(
      a1, w2, b2, nullptr, out, z, rprob, nullptr, M, 768, 3072);
}
